// Quantizer_47115791237427
// MI455X (gfx1250) — compile-verified, matrix operand order checked by reading, not run on hardware
//
#include <hip/hip_runtime.h>
#include <math.h>

typedef __attribute__((ext_vector_type(16))) __bf16 v16bf;
typedef __attribute__((ext_vector_type(8)))  float  v8f;
typedef __attribute__((ext_vector_type(4)))  float  f4v;

#define NUM_EMBS 8192
#define EMB_DIM  256
#define NVEC     16384      // B*H*W
#define ZTOT     4194304    // B*C*H*W
#define TILES_K  256        // NUM_EMBS / 32 (32 codes per iteration, 2 WMMA chains)
#define WAVES    4
#define LROW     264        // padded LDS row stride (ushorts): bank step = 4 dwords

__device__ __forceinline__ unsigned short f2bf(float f) {
    unsigned u = __float_as_uint(f);
    u += 0x7FFFu + ((u >> 16) & 1u);          // round to nearest even
    return (unsigned short)(u >> 16);
}
__device__ __forceinline__ float bf2f(unsigned short b) {
    return __uint_as_float(((unsigned)b) << 16);
}

union AFrag { v16bf v; unsigned short u[16]; };
union BFrag { v16bf v; f4v f4[2]; };

// ---------------- Kernel 1: codebook -> bf16 + |e|^2 ----------------
__global__ void k_prep_emb(const float* __restrict__ emb,
                           unsigned short* __restrict__ embb,
                           float* __restrict__ enorm) {
    const int k = blockIdx.x, d = threadIdx.x;
    float w = emb[k * EMB_DIM + d];
    unsigned short b = f2bf(w);
    embb[k * EMB_DIM + d] = b;
    float v = bf2f(b);
    __shared__ float sh[256];
    sh[d] = v * v;
    __syncthreads();
    for (int s = 128; s > 0; s >>= 1) { if (d < s) sh[d] += sh[d + s]; __syncthreads(); }
    if (d == 0) enorm[k] = sh[0];
}

// ---------------- Kernel 2: WMMA distance + argmin ----------------
__global__ void __launch_bounds__(WAVES * 32)
k_argmin(const float* __restrict__ zex, const unsigned short* __restrict__ embb,
         const float* __restrict__ enorm, int* __restrict__ idx_out) {
    __shared__ __align__(16) unsigned short lds[2][32 * LROW];

    const int tid  = threadIdx.x;
    const int wave = tid >> 5;
    const int lane = tid & 31;
    const int l15  = lane & 15;
    const int hw   = lane >> 4;
    const int mtile = blockIdx.x * WAVES + wave;
    const int row   = mtile * 16 + l15;          // x_flat row for this lane
    const int bb    = row >> 10;
    const int hwv   = row & 1023;
    const float* xbase = zex + (size_t)bb * 262144 + hwv;   // [B,C,H,W]: +c*1024 steps

    // Build A fragments (16x32 bf16 per chunk), ISA per-lane K map
    AFrag a[8];
#pragma unroll
    for (int c = 0; c < 8; ++c) {
#pragma unroll
        for (int e = 0; e < 16; ++e) {
            int k = 32 * c + 2 * ((e >> 1) & 3) + 16 * (e >> 3) + 8 * hw + (e & 1);
            a[c].u[e] = f2bf(xbase[(size_t)k * 1024]);
        }
    }

    float minv[8]; int mini[8];
#pragma unroll
    for (int v = 0; v < 8; ++v) { minv[v] = 3.4e38f; mini[v] = 0; }

    // cooperative staging: 128 threads, 128B each (32 codes x 512B per tile)
    const int srow = tid >> 2;      // 0..31
    const int sseg = tid & 3;       // 0..3
    const f4v* gsrc = (const f4v*)embb;   // 8 ushorts per f4v; 32 f4v per code row
    {
        const f4v* src = gsrc + (srow * 32 + sseg * 8);
        f4v* dst = (f4v*)&lds[0][srow * LROW + sseg * 64];
#pragma unroll
        for (int j = 0; j < 8; ++j) dst[j] = src[j];
    }
    __syncthreads();

    for (int t = 0; t < TILES_K; ++t) {
        const int cur = t & 1;
        // prefetch next tile (t=255 overshoots embb by 16KB but stays inside ws)
        f4v pf[8];
        {
            const f4v* src = gsrc + (((t + 1) * 32 + srow) * 32 + sseg * 8);
#pragma unroll
            for (int j = 0; j < 8; ++j) pf[j] = src[j];
        }

        // two independent accumulator chains: codes [t*32+l15] and [t*32+16+l15]
        v8f acc0 = {0.f,0.f,0.f,0.f,0.f,0.f,0.f,0.f};
        v8f acc1 = {0.f,0.f,0.f,0.f,0.f,0.f,0.f,0.f};
        const unsigned short* brow0 = &lds[cur][l15 * LROW + 8 * hw];
        const unsigned short* brow1 = brow0 + 16 * LROW;
#pragma unroll
        for (int c = 0; c < 8; ++c) {
            BFrag b0, b1;
            b0.f4[0] = *(const f4v*)(brow0 + 32 * c);
            b0.f4[1] = *(const f4v*)(brow0 + 32 * c + 16);
            b1.f4[0] = *(const f4v*)(brow1 + 32 * c);
            b1.f4[1] = *(const f4v*)(brow1 + 32 * c + 16);
            acc0 = __builtin_amdgcn_wmma_f32_16x16x32_bf16(
                false, a[c].v, false, b0.v, (short)0, acc0, false, false);
            acc1 = __builtin_amdgcn_wmma_f32_16x16x32_bf16(
                false, a[c].v, false, b1.v, (short)0, acc1, false, false);
        }
        const int code0 = t * 32 + l15;
        const int code1 = code0 + 16;
        const float en0 = enorm[code0];
        const float en1 = enorm[code1];
#pragma unroll
        for (int v = 0; v < 8; ++v) {
            float d0 = en0 - 2.0f * acc0[v];
            if (d0 < minv[v]) { minv[v] = d0; mini[v] = code0; }
            float d1 = en1 - 2.0f * acc1[v];
            if (d1 < minv[v]) { minv[v] = d1; mini[v] = code1; }
        }
        // stage next tile into the other buffer
        {
            f4v* dst = (f4v*)&lds[cur ^ 1][srow * LROW + sseg * 64];
#pragma unroll
            for (int j = 0; j < 8; ++j) dst[j] = pf[j];
        }
        __syncthreads();
    }

    // cross-lane argmin within each 16-lane half (C/D layout: lane = N col)
#pragma unroll
    for (int v = 0; v < 8; ++v) {
        float val = minv[v]; int ix = mini[v];
#pragma unroll
        for (int m = 8; m >= 1; m >>= 1) {
            float ov = __shfl_xor(val, m, 16);
            int   oi = __shfl_xor(ix,  m, 16);
            if (ov < val || (ov == val && oi < ix)) { val = ov; ix = oi; }
        }
        if (l15 == 0) idx_out[mtile * 16 + v + 8 * hw] = ix;
    }
}

// ---------------- Kernel 3: gather quantized, write z_q_x, SSE ----------------
__global__ void k_quant(const float* __restrict__ zex, const float* __restrict__ emb,
                        const int* __restrict__ idx, float* __restrict__ out,
                        float* __restrict__ sse) {
    const int gid = blockIdx.x * blockDim.x + threadIdx.x;   // flat over [B,C,H,W]
    const int hwv = gid & 1023;
    const int cc  = (gid >> 10) & 255;
    const int bb  = gid >> 18;
    const int n   = bb * 1024 + hwv;
    const int code = idx[n];
    const float q = emb[code * EMB_DIM + cc];
    const float x = zex[gid];
    out[1 + gid] = q;                       // z_q_x, same layout as input
    const float diff = q - x;
    __shared__ float sh[256];
    sh[threadIdx.x] = diff * diff;
    __syncthreads();
    for (int s = 128; s > 0; s >>= 1) { if (threadIdx.x < s) sh[threadIdx.x] += sh[threadIdx.x + s]; __syncthreads(); }
    if (threadIdx.x == 0) atomicAdd(sse, sh[0]);
}

// ---------------- Kernel 4: counts + idx output ----------------
__global__ void k_counts(const int* __restrict__ idx, float* __restrict__ counts,
                         float* __restrict__ out_idx) {
    const int n = blockIdx.x * blockDim.x + threadIdx.x;
    const int code = idx[n];
    atomicAdd(&counts[code], 1.0f);
    out_idx[n] = (float)code;
}

// ---------------- Kernel 5: loss + perplexity ----------------
__global__ void k_final(const float* __restrict__ counts, const float* __restrict__ sse,
                        float* __restrict__ out) {
    __shared__ float sh[256];
    float s = 0.0f;
    for (int k = threadIdx.x; k < NUM_EMBS; k += 256) {
        float p = counts[k] * (1.0f / (float)NVEC);
        s += p * logf(p + 1e-10f);
    }
    sh[threadIdx.x] = s;
    __syncthreads();
    for (int st = 128; st > 0; st >>= 1) { if (threadIdx.x < st) sh[threadIdx.x] += sh[threadIdx.x + st]; __syncthreads(); }
    if (threadIdx.x == 0) {
        out[0] = 1.25f * sse[0] / (float)ZTOT;   // loss_q + BETA*loss_e, both same MSE
        out[1 + ZTOT] = expf(-sh[0]);            // perplexity
    }
}

extern "C" void kernel_launch(void* const* d_in, const int* in_sizes, int n_in,
                              void* d_out, int out_size, void* d_ws, size_t ws_size,
                              hipStream_t stream) {
    const float* zex = (const float*)d_in[0];   // [16,256,32,32] f32
    const float* emb = (const float*)d_in[1];   // [8192,256] f32
    float* out = (float*)d_out;                 // [loss | z_q_x | perplexity | idx]
    char* ws = (char*)d_ws;

    unsigned short* embb = (unsigned short*)ws;          // 4,194,304 B
    float* enorm  = (float*)(ws + 4194304);              // 32,768 B
    float* counts = (float*)(ws + 4227072);              // 32,768 B
    float* sse    = (float*)(ws + 4259840);              // 64 B
    int*   idxb   = (int*)  (ws + 4259904);              // 65,536 B

    hipMemsetAsync(ws + 4227072, 0, 32768 + 64, stream); // zero counts + sse

    k_prep_emb<<<NUM_EMBS, 256, 0, stream>>>(emb, embb, enorm);
    k_argmin<<<NVEC / (16 * WAVES), WAVES * 32, 0, stream>>>(zex, embb, enorm, idxb);
    k_quant<<<ZTOT / 256, 256, 0, stream>>>(zex, emb, idxb, out, sse);
    k_counts<<<NVEC / 256, 256, 0, stream>>>(idxb, counts, out + 1 + ZTOT + 1);
    k_final<<<1, 256, 0, stream>>>(counts, sse, out);
}
